// TransformerBlock_24713241822031
// MI455X (gfx1250) — matrix operand order repaired and checked by reading, not compiled, not run on hardware
//
#include <hip/hip_runtime.h>

typedef __attribute__((ext_vector_type(16))) _Float16 v16h;
typedef __attribute__((ext_vector_type(8)))  float    v8f;
typedef __attribute__((ext_vector_type(4)))  int      vi4;   // clang vector int4

#define D_EMB   256
#define TSEQ    4096
#define NBATCH  4
#define BT      (NBATCH * TSEQ)   // 16384 tokens
#define DFF     512
#define LN_EPS  1e-5f

#if __has_builtin(__builtin_amdgcn_global_load_async_to_lds_b128)
#define HAS_ASYNC_LDS 1
#else
#define HAS_ASYNC_LDS 0
#endif

// ---------------------------------------------------------------------------
// WMMA helpers (gfx1250 layouts, wave32)
// ---------------------------------------------------------------------------

__device__ __forceinline__ v8f wmma_f32(v16h a, v16h b, v8f c) {
    // D = A(16x32 f16) * B(32x16 f16) + C(16x16 f32)
    return __builtin_amdgcn_wmma_f32_16x16x32_f16(
        /*neg_a=*/false, a, /*neg_b=*/false, b,
        /*c_mod=*/(short)0, c, /*reuse_a=*/false, /*reuse_b=*/false);
}

// A-tile 16x32 f16 from row-major src[M][ld] (halves).
// Lane l holds row row0+(l&15); element e holds K = (e<8?e:e+8) + 8*(l>=16).
__device__ __forceinline__ v16h load_A16x32(const _Float16* src, int row0, int ld, int kc) {
    int l  = threadIdx.x & 31;
    int hi = l >> 4;
    const _Float16* p = src + (size_t)(row0 + (l & 15)) * ld + kc + 8 * hi;
    union { v16h v; uint4 u[2]; } a;
    a.u[0] = *(const uint4*)(p);        // K = kc + 8*hi       .. +7
    a.u[1] = *(const uint4*)(p + 16);   // K = kc + 16 + 8*hi  .. +7
    return a.v;
}

// Same A-tile but source is f32; convert to f16 on the fly.
__device__ __forceinline__ v16h load_A16x32_f32(const float* src, int row0, int ld, int kc) {
    int l  = threadIdx.x & 31;
    int hi = l >> 4;
    const float* p = src + (size_t)(row0 + (l & 15)) * ld + kc + 8 * hi;
    float4 f0 = *(const float4*)(p);
    float4 f1 = *(const float4*)(p + 4);
    float4 f2 = *(const float4*)(p + 16);
    float4 f3 = *(const float4*)(p + 20);
    v16h a;
    a[0]  = (_Float16)f0.x; a[1]  = (_Float16)f0.y; a[2]  = (_Float16)f0.z; a[3]  = (_Float16)f0.w;
    a[4]  = (_Float16)f1.x; a[5]  = (_Float16)f1.y; a[6]  = (_Float16)f1.z; a[7]  = (_Float16)f1.w;
    a[8]  = (_Float16)f2.x; a[9]  = (_Float16)f2.y; a[10] = (_Float16)f2.z; a[11] = (_Float16)f2.w;
    a[12] = (_Float16)f3.x; a[13] = (_Float16)f3.y; a[14] = (_Float16)f3.z; a[15] = (_Float16)f3.w;
    return a;
}

// B-tile 32x16 f16 from src stored [N][K] row-major (weights / transposed acts,
// or an LDS-staged tile). Lane l holds column n0+(l&15);
// element e holds K = kc + (e<8?e:e+8) + 8*(l>=16).
__device__ __forceinline__ v16h load_B32x16(const _Float16* src, int n0, int ld, int kc) {
    int l  = threadIdx.x & 31;
    int hi = l >> 4;
    const _Float16* p = src + (size_t)(n0 + (l & 15)) * ld + kc + 8 * hi;
    union { v16h v; uint4 u[2]; } b;
    b.u[0] = *(const uint4*)(p);
    b.u[1] = *(const uint4*)(p + 16);
    return b.v;
}

// 16-byte global -> LDS staging copy (async on toolchains that expose it).
__device__ __forceinline__ void stage_b128(_Float16* lds_dst, const _Float16* g_src) {
#if HAS_ASYNC_LDS
    typedef __attribute__((address_space(1))) vi4 gvi4_t;   // global
    typedef __attribute__((address_space(3))) vi4 lvi4_t;   // LDS
    __builtin_amdgcn_global_load_async_to_lds_b128(
        (gvi4_t*)g_src, (lvi4_t*)lds_dst, 0, 0);
#else
    *(uint4*)lds_dst = *(const uint4*)g_src;
#endif
}

__device__ __forceinline__ void wait_async_stage() {
#if HAS_ASYNC_LDS
#if __has_builtin(__builtin_amdgcn_s_wait_asynccnt)
    __builtin_amdgcn_s_wait_asynccnt(0);
#else
    asm volatile("s_wait_asynccnt 0" ::: "memory");
#endif
#endif
}

// ---------------------------------------------------------------------------
// k0: convert qkv (with transpose), w1, w2 to f16
// ---------------------------------------------------------------------------
__global__ void k0_convert(const float* __restrict__ qkv,
                           const float* __restrict__ w1,
                           const float* __restrict__ w2,
                           _Float16* __restrict__ qkvT,
                           _Float16* __restrict__ w1h,
                           _Float16* __restrict__ w2h) {
    int idx = blockIdx.x * 256 + threadIdx.x;
    if (idx < 3 * 256 * 256) {
        int n  = idx >> 16;
        int Do = (idx >> 8) & 255;
        int d  = idx & 255;
        // qkvT[n][Do][d] = qkv[n][d][Do]  (so B-load K runs are contiguous)
        qkvT[idx] = (_Float16)qkv[(n << 16) + (d << 8) + Do];
    }
    if (idx < DFF * D_EMB) w1h[idx] = (_Float16)w1[idx];
    if (idx < D_EMB * DFF) w2h[idx] = (_Float16)w2[idx];
}

// ---------------------------------------------------------------------------
// k1: QKV projection + LN(q), LN(k); writes qh, kh [tok][256] f16 and
//     vT [b][256][4096] f16 (transposed for the P@V B-tiles).
// 512 threads = 16 waves; wave w owns output-dim tile w; block = 16 tokens.
// ---------------------------------------------------------------------------
__global__ __launch_bounds__(512) void k1_qkv_ln(
    const float* __restrict__ x, const _Float16* __restrict__ qkvT,
    const float* __restrict__ lnqw, const float* __restrict__ lnqb,
    const float* __restrict__ lnkw, const float* __restrict__ lnkb,
    _Float16* __restrict__ qh, _Float16* __restrict__ kh, _Float16* __restrict__ vT) {
    __shared__ float sq[16 * 256];
    __shared__ float sk[16 * 256];
    int wave = threadIdx.x >> 5;
    int l    = threadIdx.x & 31;
    int tok0 = blockIdx.x * 16;
    int nb   = wave * 16;

    v8f cq = {0,0,0,0,0,0,0,0}, ck = cq, cv = cq;
#pragma unroll
    for (int kc = 0; kc < D_EMB; kc += 32) {
        v16h a  = load_A16x32_f32(x, tok0, D_EMB, kc);
        v16h bq = load_B32x16(qkvT + 0 * 65536, nb, D_EMB, kc);
        v16h bk = load_B32x16(qkvT + 1 * 65536, nb, D_EMB, kc);
        v16h bv = load_B32x16(qkvT + 2 * 65536, nb, D_EMB, kc);
        cq = wmma_f32(a, bq, cq);
        ck = wmma_f32(a, bk, ck);
        cv = wmma_f32(a, bv, cv);
    }
    int hi  = l >> 4;
    int col = nb + (l & 15);
#pragma unroll
    for (int j = 0; j < 8; ++j) {
        int row = j + 8 * hi;
        sq[row * 256 + col] = cq[j];
        sk[row * 256 + col] = ck[j];
    }
    {   // v -> vT (no LN on v): 8 halves per lane are contiguous in t
        int b    = tok0 >> 12;
        int trow = (tok0 & 4095) + 8 * hi;
        union { uint4 u; _Float16 h[8]; } pk;
#pragma unroll
        for (int j = 0; j < 8; ++j) pk.h[j] = (_Float16)cv[j];
        *(uint4*)(vT + (size_t)b * (256 * 4096) + (size_t)col * 4096 + trow) = pk.u;
    }
    __syncthreads();

    // LayerNorm: wave r normalizes token-row r for both q and k
    int r = wave;
    const float* rowq = sq + r * 256;
    const float* rowk = sk + r * 256;
    int d0 = l * 8;
    float vq[8], vk[8];
    float s1q = 0.f, s1k = 0.f;
#pragma unroll
    for (int i = 0; i < 8; ++i) { vq[i] = rowq[d0 + i]; vk[i] = rowk[d0 + i]; s1q += vq[i]; s1k += vk[i]; }
    for (int m = 16; m >= 1; m >>= 1) { s1q += __shfl_xor(s1q, m, 32); s1k += __shfl_xor(s1k, m, 32); }
    float mq = s1q * (1.f / 256.f), mk = s1k * (1.f / 256.f);
    float s2q = 0.f, s2k = 0.f;
#pragma unroll
    for (int i = 0; i < 8; ++i) { float a = vq[i] - mq, b = vk[i] - mk; s2q += a * a; s2k += b * b; }
    for (int m = 16; m >= 1; m >>= 1) { s2q += __shfl_xor(s2q, m, 32); s2k += __shfl_xor(s2k, m, 32); }
    float rq = rsqrtf(s2q * (1.f / 256.f) + LN_EPS);
    float rk = rsqrtf(s2k * (1.f / 256.f) + LN_EPS);
    union { uint4 u; _Float16 h[8]; } pq, pk2;
#pragma unroll
    for (int i = 0; i < 8; ++i) {
        int d = d0 + i;
        pq.h[i]  = (_Float16)((vq[i] - mq) * rq * lnqw[d] + lnqb[d]);
        pk2.h[i] = (_Float16)((vk[i] - mk) * rk * lnkw[d] + lnkb[d]);
    }
    size_t off = (size_t)(tok0 + r) * 256 + d0;
    *(uint4*)(qh + off) = pq.u;
    *(uint4*)(kh + off) = pk2.u;
}

// ---------------------------------------------------------------------------
// k2: block-cooperative flash attention. 8 waves = 8 query tiles (128 queries,
// one batch); each 32-key K/V tile is staged once into (padded) LDS and shared
// by all waves. Probabilities bounce through per-wave LDS as f16 to re-lay the
// WMMA C-format into A-format. Epilogue: /l, LayerNorm, 0.7*(x+y) residual.
// ---------------------------------------------------------------------------
#define K2_WAVES 8
#define SK_LD 264   // 32 rows of 256 halves, padded (528B stride -> no bank conflicts)
#define SV_LD 40    // 256 rows of 32 halves, padded (80B stride)
#define SP_LD 40    // 16 rows of 32 halves, padded
__global__ __launch_bounds__(K2_WAVES * 32) void k2_attn(
    const _Float16* __restrict__ qh, const _Float16* __restrict__ kh,
    const _Float16* __restrict__ vT, const float* __restrict__ x,
    const float* __restrict__ lnaw, const float* __restrict__ lnab,
    float* __restrict__ x1f, _Float16* __restrict__ x1h) {
    __shared__ _Float16 sK[32 * SK_LD];            // staged K tile  [klocal][dim]
    __shared__ _Float16 sV[256 * SV_LD];           // staged V tile  [dim][klocal]
    __shared__ _Float16 sP[K2_WAVES * 16 * SP_LD]; // per-wave P bounce
    int wave = threadIdx.x >> 5;
    int l    = threadIdx.x & 31;
    int hi   = l >> 4;
    _Float16* myP = sP + wave * (16 * SP_LD);

    int tid  = blockIdx.x * K2_WAVES + wave;
    int tok0 = tid * 16;
    int b    = tok0 >> 12;
    int key0 = b << 12;
    const _Float16* vTb = vT + (size_t)b * (256 * 4096);

    v16h qa[8];
#pragma unroll
    for (int c = 0; c < 8; ++c) qa[c] = load_A16x32(qh, tok0, D_EMB, c * 32);

    v8f acc[16];
#pragma unroll
    for (int t = 0; t < 16; ++t) acc[t] = (v8f){0,0,0,0,0,0,0,0};
    float mrow[8], lrow[8];
#pragma unroll
    for (int j = 0; j < 8; ++j) { mrow[j] = -1e30f; lrow[j] = 0.f; }

    const float scale = 1.0f / 16.0f;   // 1/sqrt(256)

    for (int kb = 0; kb < TSEQ; kb += 32) {
        // ---- cooperative staging of K (32x256) and V (256x32) tiles ----
        const _Float16* kbase = kh + (size_t)(key0 + kb) * D_EMB;
#pragma unroll
        for (int i = 0; i < 4; ++i) {
            int p = i * 256 + (int)threadIdx.x;             // 0..1023 b128 chunks
            int kr = p >> 5, kc8 = p & 31;                  // K: row, 8-half chunk
            stage_b128(sK + kr * SK_LD + kc8 * 8, kbase + kr * 256 + kc8 * 8);
            int d = p >> 2, vc8 = p & 3;                    // V: dim row, chunk
            stage_b128(sV + d * SV_LD + vc8 * 8, vTb + (size_t)d * TSEQ + kb + vc8 * 8);
        }
        wait_async_stage();
        __syncthreads();

        // ---- S = q @ k^T (two 16-key N-subtiles) ----
        v8f s0 = (v8f){0,0,0,0,0,0,0,0}, s1 = s0;
#pragma unroll
        for (int c = 0; c < 8; ++c) {
            v16h bk0 = load_B32x16(sK, 0,  SK_LD, c * 32);
            v16h bk1 = load_B32x16(sK, 16, SK_LD, c * 32);
            s0 = wmma_f32(qa[c], bk0, s0);
            s1 = wmma_f32(qa[c], bk1, s1);
        }
        // ---- online softmax (per row j + 8*hi) ----
        float alpha[8];
#pragma unroll
        for (int j = 0; j < 8; ++j) {
            float a0 = s0[j] * scale, a1 = s1[j] * scale;
            float tm = fmaxf(a0, a1);
            for (int m = 8; m >= 1; m >>= 1) tm = fmaxf(tm, __shfl_xor(tm, m, 16));
            float mn = fmaxf(mrow[j], tm);
            float p0 = __expf(a0 - mn);
            float p1 = __expf(a1 - mn);
            float rs = p0 + p1;
            for (int m = 8; m >= 1; m >>= 1) rs += __shfl_xor(rs, m, 16);
            alpha[j] = __expf(mrow[j] - mn);
            lrow[j]  = lrow[j] * alpha[j] + rs;
            mrow[j]  = mn;
            int row = j + 8 * hi;                       // C-format -> LDS (f16)
            myP[row * SP_LD + (l & 15)]      = (_Float16)p0;
            myP[row * SP_LD + (l & 15) + 16] = (_Float16)p1;
        }
#pragma unroll
        for (int t = 0; t < 16; ++t)
#pragma unroll
            for (int j = 0; j < 8; ++j) acc[t][j] *= alpha[j];

        asm volatile("s_wait_dscnt 0" ::: "memory");
        v16h pA;                                        // A-format read-back
        {
            const _Float16* pr = myP + (l & 15) * SP_LD + 8 * hi;
            union { v16h v; uint4 u[2]; } a;
            a.u[0] = *(const uint4*)(pr);
            a.u[1] = *(const uint4*)(pr + 16);
            pA = a.v;
        }
        // ---- O += P @ V ----
#pragma unroll
        for (int t = 0; t < 16; ++t) {
            v16h bv = load_B32x16(sV, t * 16, SV_LD, 0);
            acc[t] = wmma_f32(pA, bv, acc[t]);
        }
        __syncthreads();   // before next staging overwrites sK/sV
    }

    // ---- finalize: /l, LayerNorm, residual 0.7*(x+y) ----
    float inv[8];
#pragma unroll
    for (int j = 0; j < 8; ++j) inv[j] = 1.0f / lrow[j];
    float psum[8] = {0,0,0,0,0,0,0,0};
#pragma unroll
    for (int t = 0; t < 16; ++t)
#pragma unroll
        for (int j = 0; j < 8; ++j) psum[j] += acc[t][j] * inv[j];
    float mean[8];
#pragma unroll
    for (int j = 0; j < 8; ++j) {
        float s = psum[j];
        for (int m = 8; m >= 1; m >>= 1) s += __shfl_xor(s, m, 16);
        mean[j] = s * (1.f / 256.f);
    }
    float pvar[8] = {0,0,0,0,0,0,0,0};
#pragma unroll
    for (int t = 0; t < 16; ++t)
#pragma unroll
        for (int j = 0; j < 8; ++j) { float d = acc[t][j] * inv[j] - mean[j]; pvar[j] += d * d; }
    float rstd[8];
#pragma unroll
    for (int j = 0; j < 8; ++j) {
        float s = pvar[j];
        for (int m = 8; m >= 1; m >>= 1) s += __shfl_xor(s, m, 16);
        rstd[j] = rsqrtf(s * (1.f / 256.f) + LN_EPS);
    }
#pragma unroll
    for (int t = 0; t < 16; ++t) {
        int d   = t * 16 + (l & 15);
        float w = lnaw[d], bb = lnab[d];
#pragma unroll
        for (int j = 0; j < 8; ++j) {
            size_t off = (size_t)(tok0 + j + 8 * hi) * 256 + d;
            float v  = acc[t][j] * inv[j];
            float yl = (v - mean[j]) * rstd[j] * w + bb;
            float x1 = 0.7f * (x[off] + yl);
            x1f[off] = x1;
            x1h[off] = (_Float16)x1;
        }
    }
}

// ---------------------------------------------------------------------------
// k3a: FFN1 GEMM (16384x512 = x1h @ w1^T) + leaky_relu -> h (f16).
// Each wave computes a 16x64 strip: one A load feeds 4 WMMAs.
// ---------------------------------------------------------------------------
__global__ __launch_bounds__(256) void k3a_ffn1(
    const _Float16* __restrict__ x1h, const _Float16* __restrict__ w1h,
    _Float16* __restrict__ h) {
    int wave = threadIdx.x >> 5;
    int l    = threadIdx.x & 31;
    int tile = blockIdx.x * 8 + wave;     // 8192 wave-strips
    int tok0 = (tile >> 3) * 16;          // 1024 M tiles
    int n0   = (tile & 7) * 64;           // 8 N strips of 64
    v8f c[4];
#pragma unroll
    for (int u = 0; u < 4; ++u) c[u] = (v8f){0,0,0,0,0,0,0,0};
#pragma unroll
    for (int kc = 0; kc < D_EMB; kc += 32) {
        v16h a = load_A16x32(x1h, tok0, D_EMB, kc);
#pragma unroll
        for (int u = 0; u < 4; ++u) {
            v16h bw = load_B32x16(w1h, n0 + u * 16, D_EMB, kc);
            c[u] = wmma_f32(a, bw, c[u]);
        }
    }
    int hi = l >> 4;
#pragma unroll
    for (int u = 0; u < 4; ++u) {
        int n = n0 + u * 16 + (l & 15);
#pragma unroll
        for (int j = 0; j < 8; ++j) {
            float v = c[u][j];
            v = v > 0.f ? v : 0.2f * v;
            h[(size_t)(tok0 + j + 8 * hi) * DFF + n] = (_Float16)v;
        }
    }
}

// ---------------------------------------------------------------------------
// k3b: FFN2 GEMM + leaky_relu + LN + final residual -> out (f32)
// ---------------------------------------------------------------------------
__global__ __launch_bounds__(512) void k3b_ffn2_ln(
    const _Float16* __restrict__ h, const _Float16* __restrict__ w2h,
    const float* __restrict__ x1f,
    const float* __restrict__ lnfw, const float* __restrict__ lnfb,
    float* __restrict__ out) {
    __shared__ float sy[16 * 256];
    int wave = threadIdx.x >> 5;
    int l    = threadIdx.x & 31;
    int tok0 = blockIdx.x * 16;
    int n0   = wave * 16;
    v8f c = (v8f){0,0,0,0,0,0,0,0};
#pragma unroll
    for (int kc = 0; kc < DFF; kc += 32) {
        v16h a  = load_A16x32(h, tok0, DFF, kc);
        v16h bw = load_B32x16(w2h, n0, DFF, kc);
        c = wmma_f32(a, bw, c);
    }
    int hi = l >> 4;
    int n  = n0 + (l & 15);
#pragma unroll
    for (int j = 0; j < 8; ++j) {
        float v = c[j];
        v = v > 0.f ? v : 0.2f * v;
        sy[(j + 8 * hi) * 256 + n] = v;
    }
    __syncthreads();
    int r  = wave;
    float* row = sy + r * 256;
    int d0 = l * 8;
    float vv[8]; float s = 0.f;
#pragma unroll
    for (int i = 0; i < 8; ++i) { vv[i] = row[d0 + i]; s += vv[i]; }
    for (int m = 16; m >= 1; m >>= 1) s += __shfl_xor(s, m, 32);
    float mean = s * (1.f / 256.f);
    float s2 = 0.f;
#pragma unroll
    for (int i = 0; i < 8; ++i) { float d = vv[i] - mean; s2 += d * d; }
    for (int m = 16; m >= 1; m >>= 1) s2 += __shfl_xor(s2, m, 32);
    float rstd = rsqrtf(s2 * (1.f / 256.f) + LN_EPS);
    size_t off = (size_t)(tok0 + r) * 256 + d0;
    float ov[8];
#pragma unroll
    for (int i = 0; i < 8; ++i) {
        int d = d0 + i;
        float yl = (vv[i] - mean) * rstd * lnfw[d] + lnfb[d];
        ov[i] = 0.7f * (x1f[off + i] + yl);
    }
    *(float4*)(out + off)     = make_float4(ov[0], ov[1], ov[2], ov[3]);
    *(float4*)(out + off + 4) = make_float4(ov[4], ov[5], ov[6], ov[7]);
}

// ---------------------------------------------------------------------------
// launch
// ---------------------------------------------------------------------------
extern "C" void kernel_launch(void* const* d_in, const int* in_sizes, int n_in,
                              void* d_out, int out_size, void* d_ws, size_t ws_size,
                              hipStream_t stream) {
    const float* x    = (const float*)d_in[0];
    const float* qkv  = (const float*)d_in[1];
    const float* lnqw = (const float*)d_in[2];
    const float* lnqb = (const float*)d_in[3];
    const float* lnkw = (const float*)d_in[4];
    const float* lnkb = (const float*)d_in[5];
    const float* lnaw = (const float*)d_in[6];
    const float* lnab = (const float*)d_in[7];
    const float* w1   = (const float*)d_in[8];
    const float* w2   = (const float*)d_in[9];
    const float* lnfw = (const float*)d_in[10];
    const float* lnfb = (const float*)d_in[11];
    float* out = (float*)d_out;

    char* ws = (char*)d_ws;
    size_t o = 0;
    _Float16* qh    = (_Float16*)(ws + o); o += (size_t)BT * D_EMB * 2;       // 8 MB
    _Float16* kh    = (_Float16*)(ws + o); o += (size_t)BT * D_EMB * 2;       // 8 MB
    _Float16* vT    = (_Float16*)(ws + o); o += (size_t)BT * D_EMB * 2;       // 8 MB
    float*    x1f   = (float*)   (ws + o); o += (size_t)BT * D_EMB * 4;       // 16 MB
    _Float16* x1h   = (_Float16*)(ws + o); o += (size_t)BT * D_EMB * 2;       // 8 MB
    _Float16* hbuf  = (_Float16*)(ws + o); o += (size_t)BT * DFF * 2;         // 16 MB
    _Float16* qkvT  = (_Float16*)(ws + o); o += (size_t)3 * D_EMB * D_EMB * 2;
    _Float16* w1h   = (_Float16*)(ws + o); o += (size_t)DFF * D_EMB * 2;
    _Float16* w2h   = (_Float16*)(ws + o); o += (size_t)D_EMB * DFF * 2;

    k0_convert<<<768, 256, 0, stream>>>(qkv, w1, w2, qkvT, w1h, w2h);
    k1_qkv_ln<<<BT / 16, 512, 0, stream>>>(x, qkvT, lnqw, lnqb, lnkw, lnkb, qh, kh, vT);
    k2_attn<<<(BT / 16) / K2_WAVES, K2_WAVES * 32, 0, stream>>>(qh, kh, vT, x, lnaw, lnab, x1f, x1h);
    k3a_ffn1<<<(BT / 16) * (DFF / 64) / 8, 256, 0, stream>>>(x1h, w1h, hbuf);
    k3b_ffn2_ln<<<BT / 16, 512, 0, stream>>>(hbuf, w2h, x1f, lnfw, lnfb, out);
}
